// STGAAT_17085379903856
// MI455X (gfx1250) — matrix operand order repaired and checked by reading, not compiled, not run on hardware
//
#include <hip/hip_runtime.h>
#include <hip/hip_bf16.h>
#include <math.h>

// Problem constants (from reference)
#define B_ 16
#define T_ 12
#define N_ 512
#define FI_ 2
#define FO_ 32
#define H_ 64
#define NEG_ (-9000000000000000.0f)
#define SLOPE_ 0.01f

typedef __attribute__((ext_vector_type(16))) __bf16 v16bf;
typedef __attribute__((ext_vector_type(8)))  float  v8f;

union Frag16 { uint4 q[2]; v16bf v; };
union Pack8  { unsigned short u[8]; uint4 q; };

__device__ __forceinline__ unsigned short f2bf(float f) {
  unsigned int u = __float_as_uint(f);
  u += 0x7FFFu + ((u >> 16) & 1u);           // round-to-nearest-even
  return (unsigned short)(u >> 16);
}
__device__ __forceinline__ float bf2f(unsigned short h) {
  return __uint_as_float(((unsigned int)h) << 16);
}
__device__ __forceinline__ float sigf(float x) { return 1.0f / (1.0f + __expf(-x)); }

// ---------------------------------------------------------------------------
// Kernel 0: bit-pack adjacency: adjp[row*16 + w] bit k = (adj[row, w*32+k] > 0)
// ---------------------------------------------------------------------------
__global__ __launch_bounds__(256) void k0_pack(const int* __restrict__ adj,
                                               unsigned int* __restrict__ adjp) {
  int t = blockIdx.x * 256 + threadIdx.x;    // covers 512*512
  int pred = adj[t] > 0 ? 1 : 0;
  unsigned long long b = __ballot(pred);     // wave32: low 32 bits valid
  if ((threadIdx.x & 31) == 0) adjp[t >> 5] = (unsigned int)b;
}

// ---------------------------------------------------------------------------
// Kernel 1: Wh projection (FI=2 -> FO=32), stored transposed [b,t][o][n] bf16
// (B-operand layout for the attention WMMA), plus s1/s2 = Wh @ a1/a2 in fp32.
// ---------------------------------------------------------------------------
__global__ __launch_bounds__(256) void k1_project(
    const float* __restrict__ x, const float* __restrict__ Wg,
    const float* __restrict__ av, unsigned short* __restrict__ WhT,
    float* __restrict__ s1g, float* __restrict__ s2g) {
  int r = blockIdx.x * blockDim.x + threadIdx.x;
  if (r >= B_ * T_ * N_) return;
  float x0 = x[2 * r], x1 = x[2 * r + 1];
  int bt = r >> 9, n = r & (N_ - 1);
  float s1 = 0.f, s2 = 0.f;
#pragma unroll
  for (int o = 0; o < FO_; ++o) {
    float wh = x0 * Wg[2 * o] + x1 * Wg[2 * o + 1];
    s1 += wh * av[o];
    s2 += wh * av[FO_ + o];
    WhT[(size_t)bt * (FO_ * N_) + o * N_ + n] = f2bf(wh);
  }
  s1g[r] = s1;
  s2g[r] = s2;
}

// ---------------------------------------------------------------------------
// Kernel 2: fused GAT attention per (b,t). Branchless masked softmax on the
// bit-packed adjacency. Each lane computes exactly the 16 exp values its
// WMMA A-fragment needs (two 8-col runs), so the exp tile lands directly in
// fragment registers -- no shuffles, no LDS round trip in the GEMM loop.
// ---------------------------------------------------------------------------
__global__ __launch_bounds__(256) void k2_gat(
    const unsigned int* __restrict__ adjp, const unsigned short* __restrict__ WhT,
    const float* __restrict__ s1g, const float* __restrict__ s2g,
    unsigned short* __restrict__ hseq) {
  __shared__ unsigned short sWhT[FO_][N_ + 8];  // [o][j] bf16, padded rows
  __shared__ float sS2[N_];
  __shared__ float sDen[8][16];

  int bt = blockIdx.x;  // b*T + t
  int b = bt / T_, t = bt % T_;
  int tid = threadIdx.x;

  for (int e = tid; e < FO_ * N_; e += 256) {
    int o = e >> 9, n = e & (N_ - 1);
    sWhT[o][n] = WhT[(size_t)bt * FO_ * N_ + e];
  }
  for (int n = tid; n < N_; n += 256) sS2[n] = s2g[bt * N_ + n];
  __syncthreads();

  const float4* sS24 = (const float4*)sS2;
  int w = tid >> 5, lane = tid & 31;
  int r = lane & 15, half = lane >> 4;

  for (int it = w; it < N_ / 16; it += 8) {
    int i0 = it * 16;
    int irow = i0 + r;
    float s1i = s1g[bt * N_ + irow];
    const unsigned int* arow = adjp + irow * 16;

    // pass 1: per-row max of masked leaky-relu scores (branchless).
    // lane covers cols {half*8+0..7} and {16+half*8+0..7} of each 32-chunk;
    // the lane pair (l, l^16) covers the full chunk.
    float m = NEG_;
    for (int ci = 0; ci < 16; ++ci) {
      unsigned int word = arow[ci];
      unsigned int mk0 = (word >> (half * 8)) & 0xffu;
      unsigned int mk1 = (word >> (16 + half * 8)) & 0xffu;
      int qb = ci * 8 + half * 2;
      auto mx = [&](float sval, unsigned int mk, int k) {
        float e = s1i + sval;
        e = e > 0.f ? e : SLOPE_ * e;
        m = fmaxf(m, ((mk >> k) & 1u) ? e : NEG_);
      };
#pragma unroll
      for (int q = 0; q < 2; ++q) {
        float4 sv = sS24[qb + q];
        mx(sv.x, mk0, q * 4 + 0); mx(sv.y, mk0, q * 4 + 1);
        mx(sv.z, mk0, q * 4 + 2); mx(sv.w, mk0, q * 4 + 3);
      }
#pragma unroll
      for (int q = 0; q < 2; ++q) {
        float4 sv = sS24[qb + 4 + q];
        mx(sv.x, mk1, q * 4 + 0); mx(sv.y, mk1, q * 4 + 1);
        mx(sv.z, mk1, q * 4 + 2); mx(sv.w, mk1, q * 4 + 3);
      }
    }
    m = fmaxf(m, __shfl_xor(m, 16));

    // pass 2: exp runs -> A fragment registers -> WMMA accumulate att @ Wh
    v8f c0 = {}, c1 = {};
    float den = 0.f;
    for (int jt = 0; jt < N_ / 32; ++jt) {
      unsigned int word = arow[jt];
      unsigned int mk0 = (word >> (half * 8)) & 0xffu;
      unsigned int mk1 = (word >> (16 + half * 8)) & 0xffu;
      int qb = jt * 8 + half * 2;
      Pack8 p0, p1;
      auto ex = [&](float sval, unsigned int mk, int k, Pack8& pp) {
        float e = s1i + sval;
        e = e > 0.f ? e : SLOPE_ * e;
        e = ((mk >> k) & 1u) ? e : NEG_;
        float p = __expf(e - m);
        den += p;
        pp.u[k] = f2bf(p);
      };
#pragma unroll
      for (int q = 0; q < 2; ++q) {
        float4 sv = sS24[qb + q];
        ex(sv.x, mk0, q * 4 + 0, p0); ex(sv.y, mk0, q * 4 + 1, p0);
        ex(sv.z, mk0, q * 4 + 2, p0); ex(sv.w, mk0, q * 4 + 3, p0);
      }
#pragma unroll
      for (int q = 0; q < 2; ++q) {
        float4 sv = sS24[qb + 4 + q];
        ex(sv.x, mk1, q * 4 + 0, p1); ex(sv.y, mk1, q * 4 + 1, p1);
        ex(sv.z, mk1, q * 4 + 2, p1); ex(sv.w, mk1, q * 4 + 3, p1);
      }
      Frag16 af;
      af.q[0] = p0.q;   // K = half*8 + 0..7
      af.q[1] = p1.q;   // K = 16 + half*8 + 0..7
      // B fragments: Wh[j][o] from transposed LDS (lane = col o, K contiguous)
      Frag16 bf0, bf1;
      bf0.q[0] = *(const uint4*)(&sWhT[r][jt * 32 + half * 8]);
      bf0.q[1] = *(const uint4*)(&sWhT[r][jt * 32 + 16 + half * 8]);
      bf1.q[0] = *(const uint4*)(&sWhT[16 + r][jt * 32 + half * 8]);
      bf1.q[1] = *(const uint4*)(&sWhT[16 + r][jt * 32 + 16 + half * 8]);
      c0 = __builtin_amdgcn_wmma_f32_16x16x32_bf16(false, af.v, false, bf0.v,
                                                   (short)0, c0, false, false);
      c1 = __builtin_amdgcn_wmma_f32_16x16x32_bf16(false, af.v, false, bf1.v,
                                                   (short)0, c1, false, false);
    }
    den += __shfl_xor(den, 16);
    sDen[w][r] = den;   // both halves hold identical value: benign same-value write

#pragma unroll
    for (int vr = 0; vr < 8; ++vr) {
      int row = vr + 8 * half;  // C-fragment M index
      float dv = sDen[w][row];
      float inv = dv > 0.f ? 1.0f / dv : 0.f;
      size_t base = ((size_t)(b * N_ + i0 + row) * T_ + t) * FO_;
      hseq[base + r]      = f2bf(fmaxf(c0[vr] * inv, 0.f));
      hseq[base + 16 + r] = f2bf(fmaxf(c1[vr] * inv, 0.f));
    }
  }
}

// ---------------------------------------------------------------------------
// Kernel 3: batched LSTM (8192 rows x T=12, H=64) + final 64->1 projection.
// Each wave owns 16 rows; h/c live in fp32 C-fragments across all steps;
// weights staged once in LDS as bf16 in B-fragment layout. 48 WMMAs/step/wave.
// ---------------------------------------------------------------------------
__global__ __launch_bounds__(128) void k3_lstm(
    const unsigned short* __restrict__ hseq, const float* __restrict__ w_ih,
    const float* __restrict__ w_hh, const float* __restrict__ b_ih,
    const float* __restrict__ b_hh, const float* __restrict__ W_fc,
    const float* __restrict__ b_fc, float* __restrict__ out) {
  __shared__ unsigned short sWih[4 * H_ * FO_];  // [256][32] bf16
  __shared__ unsigned short sWhh[4 * H_ * H_];   // [256][64] bf16
  __shared__ unsigned short sH[4][16 * H_];      // per-wave h staging [16][64]

  int tid = threadIdx.x;
  for (int e = tid; e < 4 * H_ * FO_; e += 128) sWih[e] = f2bf(w_ih[e]);
  for (int e = tid; e < 4 * H_ * H_; e += 128) sWhh[e] = f2bf(w_hh[e]);
  __syncthreads();

  int w = tid >> 5, lane = tid & 31;
  int r = lane & 15, half = lane >> 4;
  int bn0 = blockIdx.x * 64 + w * 16;

  // hoist biases (per-lane column gt*16 + r)
  float bias[4][4];
#pragma unroll
  for (int nt = 0; nt < 4; ++nt)
#pragma unroll
    for (int gg = 0; gg < 4; ++gg) {
      int gcol = (nt + 4 * gg) * 16 + r;
      bias[nt][gg] = b_ih[gcol] + b_hh[gcol];
    }

  v8f hfr[4] = {}, cfr[4] = {};

  for (int t = 0; t < T_; ++t) {
    // stage current h (fragment layout) into row-major bf16 [16][64]
#pragma unroll
    for (int nt = 0; nt < 4; ++nt)
#pragma unroll
      for (int vr = 0; vr < 8; ++vr) {
        int row = vr + 8 * half;
        sH[w][row * H_ + nt * 16 + r] = f2bf(hfr[nt][vr]);
      }
    // A fragment: x_t tile straight from global (64B rows, 16B aligned)
    Frag16 ax;
    {
      const unsigned short* xrow = hseq + ((size_t)(bn0 + r) * T_ + t) * FO_;
      ax.q[0] = *(const uint4*)(xrow + half * 8);
      ax.q[1] = *(const uint4*)(xrow + 16 + half * 8);
    }
    // A fragments: h split into two K=32 tiles
    Frag16 ah1, ah2;
    {
      const unsigned short* hrow = &sH[w][r * H_];
      ah1.q[0] = *(const uint4*)(hrow + half * 8);
      ah1.q[1] = *(const uint4*)(hrow + 16 + half * 8);
      ah2.q[0] = *(const uint4*)(hrow + 32 + half * 8);
      ah2.q[1] = *(const uint4*)(hrow + 48 + half * 8);
    }
#pragma unroll
    for (int nt = 0; nt < 4; ++nt) {
      v8f g[4];
#pragma unroll
      for (int gg = 0; gg < 4; ++gg) {
        int gcol = (nt + 4 * gg) * 16 + r;   // gate tiles: i=0..3 f=4..7 g=8..11 o=12..15
        Frag16 bih, bh1, bh2;
        const unsigned short* wr = &sWih[gcol * FO_];
        bih.q[0] = *(const uint4*)(wr + half * 8);
        bih.q[1] = *(const uint4*)(wr + 16 + half * 8);
        const unsigned short* hr = &sWhh[gcol * H_];
        bh1.q[0] = *(const uint4*)(hr + half * 8);
        bh1.q[1] = *(const uint4*)(hr + 16 + half * 8);
        bh2.q[0] = *(const uint4*)(hr + 32 + half * 8);
        bh2.q[1] = *(const uint4*)(hr + 48 + half * 8);
        v8f acc = {};
        acc = __builtin_amdgcn_wmma_f32_16x16x32_bf16(false, ax.v, false, bih.v,
                                                      (short)0, acc, false, false);
        acc = __builtin_amdgcn_wmma_f32_16x16x32_bf16(false, ah1.v, false, bh1.v,
                                                      (short)0, acc, false, false);
        acc = __builtin_amdgcn_wmma_f32_16x16x32_bf16(false, ah2.v, false, bh2.v,
                                                      (short)0, acc, false, false);
        g[gg] = acc;
      }
      // elementwise LSTM cell update directly on C-fragment layout
#pragma unroll
      for (int vr = 0; vr < 8; ++vr) {
        float iv = sigf(g[0][vr] + bias[nt][0]);
        float fv = sigf(g[1][vr] + bias[nt][1]);
        float gv = tanhf(g[2][vr] + bias[nt][2]);
        float ov = sigf(g[3][vr] + bias[nt][3]);
        float c = fv * cfr[nt][vr] + iv * gv;
        cfr[nt][vr] = c;
        hfr[nt][vr] = ov * tanhf(c);
      }
    }
  }

  // final projection: out[bn] = h_last . W_fc + b_fc
#pragma unroll
  for (int nt = 0; nt < 4; ++nt)
#pragma unroll
    for (int vr = 0; vr < 8; ++vr) {
      int row = vr + 8 * half;
      sH[w][row * H_ + nt * 16 + r] = f2bf(hfr[nt][vr]);
    }
  float acc = 0.f;
#pragma unroll
  for (int k = 0; k < 32; ++k) {
    int c = half * 32 + k;
    acc += bf2f(sH[w][r * H_ + c]) * W_fc[c];
  }
  acc += __shfl_xor(acc, 16);
  out[bn0 + r] = acc + b_fc[0];   // both halves identical: benign same-value write
}

// ---------------------------------------------------------------------------
extern "C" void kernel_launch(void* const* d_in, const int* in_sizes, int n_in,
                              void* d_out, int out_size, void* d_ws,
                              size_t ws_size, hipStream_t stream) {
  const float* x     = (const float*)d_in[0];
  const int*   adj   = (const int*)d_in[1];
  const float* W_gat = (const float*)d_in[2];
  const float* a_vec = (const float*)d_in[3];
  const float* w_ih  = (const float*)d_in[4];
  const float* w_hh  = (const float*)d_in[5];
  const float* b_ih  = (const float*)d_in[6];
  const float* b_hh  = (const float*)d_in[7];
  const float* W_fc  = (const float*)d_in[8];
  const float* b_fc  = (const float*)d_in[9];
  float* out = (float*)d_out;

  char* ws = (char*)d_ws;
  unsigned short* WhT = (unsigned short*)ws;               // 6,291,456 B
  float* s1 = (float*)(ws + 6291456);                      //   393,216 B
  float* s2 = (float*)(ws + 6684672);                      //   393,216 B
  unsigned short* hseq = (unsigned short*)(ws + 7077888);  // 6,291,456 B
  unsigned int* adjp = (unsigned int*)(ws + 13369344);     //    32,768 B

  k0_pack<<<(N_ * N_) / 256, 256, 0, stream>>>(adj, adjp);
  k1_project<<<(B_ * T_ * N_) / 256, 256, 0, stream>>>(x, W_gat, a_vec, WhT, s1, s2);
  k2_gat<<<B_ * T_, 256, 0, stream>>>(adjp, WhT, s1, s2, hseq);
  k3_lstm<<<(B_ * N_) / 64, 128, 0, stream>>>(hseq, w_ih, w_hh, b_ih, b_hh, W_fc, b_fc, out);
}
